// RBM_4097398801002
// MI455X (gfx1250) — matrix operand order repaired and checked by reading, not compiled, not run on hardware
//
#include <hip/hip_runtime.h>
#include <cstdint>

typedef __attribute__((ext_vector_type(16))) _Float16 v16h;
typedef __attribute__((ext_vector_type(8)))  float    v8f;

#define D_DIM          256
#define STEPS          10
#define NTHREADS       128   // 4 waves
#define ROWS_PER_WAVE  32    // two 16-row WMMA tiles per wave
#define ROWS_PER_BLOCK 128   // 4 waves * 32 rows
#define ZSTR           264   // padded z row stride (halfs)
#define WSTR           264   // padded Wt row stride (halfs)

__device__ __forceinline__ uint32_t mix32(uint32_t x) {
  x ^= x >> 16; x *= 0x7FEB352Du;
  x ^= x >> 15; x *= 0x846CA68Bu;
  x ^= x >> 16;
  return x;
}
// Cheap per-thread stream: 2 VALU ops per draw, deterministic in program order.
__device__ __forceinline__ float lcg_u01(uint32_t& s) {
  s = s * 1664525u + 1013904223u;
  return (float)(s >> 8) * (1.0f / 16777216.0f);
}

union Frag { v16h h; uint32_t u[8]; };

__global__ __launch_bounds__(NTHREADS, 1)
void rbm_gibbs_energy(const float* __restrict__ hvec,
                      const float* __restrict__ W,
                      float* __restrict__ energy)
{
  // Wt[n][k] = (f16)W[k][n]: B-fragments are K-contiguous packed f16 pairs.
  __shared__ _Float16 Wt[D_DIM * WSTR];
  __shared__ _Float16 Zb[2][ROWS_PER_BLOCK * ZSTR];
  __shared__ float    hs[D_DIM];

  const int tid  = threadIdx.x;
  const int lane = tid & 31;
  const int wave = tid >> 5;
  const int nloc = lane & 15;              // N within 16x16 tile (B/C layout)
  const int hi   = lane >> 4;              // lane half
  const int rowbase = wave * ROWS_PER_WAVE;
  const int sample_base = blockIdx.x * ROWS_PER_BLOCK;

  uint32_t rng = mix32((uint32_t)(blockIdx.x * NTHREADS + tid) * 0x9E3779B1u + 0x85EBCA77u);

  // ---- stage h, and W (f32 -> f16, transposed) into LDS ----
  hs[tid]       = hvec[tid];
  hs[tid + 128] = hvec[tid + 128];
  #pragma unroll 4
  for (int k = 0; k < D_DIM; ++k) {
    Wt[tid * WSTR + k]         = (_Float16)W[k * D_DIM + tid];
    Wt[(tid + 128) * WSTR + k] = (_Float16)W[k * D_DIM + tid + 128];
  }

  // ---- z0 ~ Bernoulli(0.5) ----
  for (int e = tid; e < ROWS_PER_BLOCK * D_DIM; e += NTHREADS) {
    const int r = e >> 8, c = e & 255;
    rng = rng * 1664525u + 1013904223u;
    Zb[0][r * ZSTR + c] = (rng >> 31) ? (_Float16)1.0f : (_Float16)0.0f;
  }
  __syncthreads();

  int cur = 0;

  #pragma unroll 1
  for (int step = 0; step < STEPS; ++step) {
    // Two 16x256 z tiles -> 2 x 8 f16 A-fragments (ISA 16-bit A layout:
    // M = lane&15; VGPR i holds packed K = (i>>2)*16 + hi*8 + 2*(i&3)).
    Frag A0[8], A1[8];
    #pragma unroll
    for (int kt = 0; kt < 8; ++kt) {
      #pragma unroll
      for (int i = 0; i < 8; ++i) {
        const int klo = ((i >> 2) << 4) + (hi << 3) + ((i & 3) << 1);
        A0[kt].u[i] = *(const uint32_t*)&Zb[cur][(rowbase + nloc) * ZSTR + kt * 32 + klo];
        A1[kt].u[i] = *(const uint32_t*)&Zb[cur][(rowbase + 16 + nloc) * ZSTR + kt * 32 + klo];
      }
    }
    #pragma unroll 1
    for (int j = 0; j < 16; ++j) {
      v8f acc0 = {}, acc1 = {};
      #pragma unroll
      for (int kt = 0; kt < 8; ++kt) {
        // B layout: N = lane&15; VGPR i holds packed K = (i>>2)*16 + hi*8 + 2*(i&3)
        Frag B;
        #pragma unroll
        for (int i = 0; i < 8; ++i) {
          const int klo = ((i >> 2) << 4) + (hi << 3) + ((i & 3) << 1);
          B.u[i] = *(const uint32_t*)&Wt[(j * 16 + nloc) * WSTR + kt * 32 + klo];
        }
        acc0 = __builtin_amdgcn_wmma_f32_16x16x32_f16(false, A0[kt].h, false, B.h,
                                                      (short)0, acc0, false, false);
        acc1 = __builtin_amdgcn_wmma_f32_16x16x32_f16(false, A1[kt].h, false, B.h,
                                                      (short)0, acc1, false, false);
      }
      // Gibbs update; C layout: m = r + 8*hi, n = nloc. hs[col] invariant in r.
      const int col = j * 16 + nloc;
      const float hval = hs[col];
      #pragma unroll
      for (int r = 0; r < 8; ++r) {
        const int m = r + (hi << 3);
        // tile 0
        {
          const float e  = __expf(-(acc0[r] + hval));
          const float u  = lcg_u01(rng);
          const bool  on = (u + u * e) < 1.0f;   // u < sigmoid(act)
          Zb[cur ^ 1][(rowbase + m) * ZSTR + col] = on ? (_Float16)1.0f : (_Float16)0.0f;
        }
        // tile 1
        {
          const float e  = __expf(-(acc1[r] + hval));
          const float u  = lcg_u01(rng);
          const bool  on = (u + u * e) < 1.0f;
          Zb[cur ^ 1][(rowbase + 16 + m) * ZSTR + col] = on ? (_Float16)1.0f : (_Float16)0.0f;
        }
      }
    }
    cur ^= 1;
    // No barrier: each wave reads/writes only its own 32 z rows; same-wave
    // LDS ops execute in order (DScnt), Wt/hs are read-only after init.
  }

  // ---- energy pass: e[m] = sum_col z[m][col] * (h[col] + (zW)[m][col]) ----
  float ep0[8] = {0.f,0.f,0.f,0.f,0.f,0.f,0.f,0.f};
  float ep1[8] = {0.f,0.f,0.f,0.f,0.f,0.f,0.f,0.f};
  {
    Frag A0[8], A1[8];
    #pragma unroll
    for (int kt = 0; kt < 8; ++kt) {
      #pragma unroll
      for (int i = 0; i < 8; ++i) {
        const int klo = ((i >> 2) << 4) + (hi << 3) + ((i & 3) << 1);
        A0[kt].u[i] = *(const uint32_t*)&Zb[cur][(rowbase + nloc) * ZSTR + kt * 32 + klo];
        A1[kt].u[i] = *(const uint32_t*)&Zb[cur][(rowbase + 16 + nloc) * ZSTR + kt * 32 + klo];
      }
    }
    #pragma unroll 1
    for (int j = 0; j < 16; ++j) {
      v8f acc0 = {}, acc1 = {};
      #pragma unroll
      for (int kt = 0; kt < 8; ++kt) {
        Frag B;
        #pragma unroll
        for (int i = 0; i < 8; ++i) {
          const int klo = ((i >> 2) << 4) + (hi << 3) + ((i & 3) << 1);
          B.u[i] = *(const uint32_t*)&Wt[(j * 16 + nloc) * WSTR + kt * 32 + klo];
        }
        acc0 = __builtin_amdgcn_wmma_f32_16x16x32_f16(false, A0[kt].h, false, B.h,
                                                      (short)0, acc0, false, false);
        acc1 = __builtin_amdgcn_wmma_f32_16x16x32_f16(false, A1[kt].h, false, B.h,
                                                      (short)0, acc1, false, false);
      }
      const int col = j * 16 + nloc;
      const float hval = hs[col];
      #pragma unroll
      for (int r = 0; r < 8; ++r) {
        const int m = r + (hi << 3);
        const float zv0 = (float)Zb[cur][(rowbase + m) * ZSTR + col];
        const float zv1 = (float)Zb[cur][(rowbase + 16 + m) * ZSTR + col];
        ep0[r] += zv0 * (hval + acc0[r]);
        ep1[r] += zv1 * (hval + acc1[r]);
      }
    }
  }
  // Reduce across lanes 0-3 bits (N columns); bit 4 (row-group) untouched.
  #pragma unroll
  for (int r = 0; r < 8; ++r) {
    float v0 = ep0[r], v1 = ep1[r];
    v0 += __shfl_xor(v0, 1, 32);  v1 += __shfl_xor(v1, 1, 32);
    v0 += __shfl_xor(v0, 2, 32);  v1 += __shfl_xor(v1, 2, 32);
    v0 += __shfl_xor(v0, 4, 32);  v1 += __shfl_xor(v1, 4, 32);
    v0 += __shfl_xor(v0, 8, 32);  v1 += __shfl_xor(v1, 8, 32);
    ep0[r] = v0; ep1[r] = v1;
  }
  if (nloc == 0) {
    #pragma unroll
    for (int r = 0; r < 8; ++r) {
      const int m = r + (hi << 3);
      energy[sample_base + rowbase + m]      = -ep0[r];
      energy[sample_base + rowbase + 16 + m] = -ep1[r];
    }
  }
}

extern "C" void kernel_launch(void* const* d_in, const int* in_sizes, int n_in,
                              void* d_out, int out_size, void* d_ws, size_t ws_size,
                              hipStream_t stream) {
  const float* h = (const float*)d_in[0];
  const float* W = (const float*)d_in[1];
  float* energy  = (float*)d_out;
  (void)in_sizes; (void)n_in; (void)d_ws; (void)ws_size;
  const int nsamp = out_size;                   // 131072
  const int grid  = nsamp / ROWS_PER_BLOCK;     // 1024 blocks of 4 waves
  rbm_gibbs_energy<<<grid, NTHREADS, 0, stream>>>(h, W, energy);
}
